// CRF_18605798326644
// MI455X (gfx1250) — matrix operand order repaired and checked by reading, not compiled, not run on hardware
//
#include <hip/hip_runtime.h>
#include <hip/hip_bf16.h>

#define Bsz 512
#define Tlen 512
#define NL 128
#define BOS_LAB 1
#define EOS_LAB 2

typedef __bf16 bf16_t;
typedef __attribute__((ext_vector_type(16))) __bf16 v16bf;
typedef __attribute__((ext_vector_type(8)))  float  v8f;

union V16U { v16bf v; uint4 q[2]; };

static __device__ inline unsigned pack2bf(bf16_t a, bf16_t b) {
    union { bf16_t h; unsigned short u; } ua, ub;
    ua.h = a; ub.h = b;
    return (unsigned)ua.u | ((unsigned)ub.u << 16);
}

// ---------------------------------------------------------------------------
// Kernel 1: gold path score, one wave per batch row.
// ---------------------------------------------------------------------------
__global__ __launch_bounds__(256) void crf_gold(
    const float* __restrict__ feat, const int* __restrict__ tags,
    const unsigned char* __restrict__ mask, const float* __restrict__ tr,
    float* __restrict__ gold)
{
    int gid  = blockIdx.x * blockDim.x + threadIdx.x;
    int b    = gid >> 5;
    int lane = threadIdx.x & 31;
    if (b >= Bsz) return;

    const int* tb = tags + b * Tlen;
    const unsigned char* mb = mask + b * Tlen;

    float acc = 0.0f;
    int   cnt = 0;
    for (int t = lane; t < Tlen; t += 32) {
        int tg = tb[t];
        int mk = (int)mb[t];
        cnt += mk;
        if (t >= 1 && mk) {
            int tp = tb[t - 1];
            acc += feat[(size_t)b * Tlen * NL + (size_t)t * NL + tg]
                 + tr[tp * NL + tg];
        }
    }
    for (int d = 16; d >= 1; d >>= 1) {
        acc += __shfl_xor(acc, d, 32);
        cnt += __shfl_xor(cnt, d, 32);
    }
    if (lane == 0) {
        int t0 = tb[0];
        float g = acc + feat[(size_t)b * Tlen * NL + t0] + tr[BOS_LAB * NL + t0];
        int vlen  = cnt - 1;
        int lastl = tb[vlen];
        g += tr[lastl * NL + EOS_LAB];
        gold[b] = g;
    }
}

// ---------------------------------------------------------------------------
// Kernel 2: forward algorithm. 4 waves/block, each wave owns 16 batch rows.
// State: fv[b,j] = Mcum[b] + log(Sv[b,j]), Sv linear:
//   maxval = max_j Sv ; P = Sv/maxval ; S = P @ exp(tr) (bf16 hi/lo WMMA) ;
//   Sv' = S * exp(feat_t) ; Mcum += log(maxval)
// ---------------------------------------------------------------------------
__global__ __launch_bounds__(128) void crf_forward(
    const float* __restrict__ feat, const unsigned char* __restrict__ mask,
    const float* __restrict__ tr, float* __restrict__ logZ)
{
    // E matrices in B-operand lane layout, K-permuted (i = (K%8)*16 + K/8).
    __shared__ __attribute__((aligned(32))) bf16_t EBhi[4 * 8 * 32 * 16];
    __shared__ __attribute__((aligned(32))) bf16_t EBlo[4 * 8 * 32 * 16];
    // Per-wave P staging (row-major 16x128, columns permuted: col = c*8 + nt).
    __shared__ __attribute__((aligned(32))) bf16_t PAhi[4][16 * NL];
    __shared__ __attribute__((aligned(32))) bf16_t PAlo[4][16 * NL];

    const int tid     = threadIdx.x;
    const int wave    = tid >> 5;
    const int lane    = tid & 31;
    const int c       = lane & 15;   // column-in-tile
    const int halfSel = lane >> 4;   // 0 or 1
    const int b0      = blockIdx.x * 64 + wave * 16;

    // ---- Precompute E = exp(tr) as bf16 hi/lo in B-operand layout ----
    for (int s = tid; s < 4 * 8 * 32 * 16; s += 128) {
        int e  = s & 15;
        int ln = (s >> 4) & 31;
        int nt = (s >> 9) & 7;
        int kc = (s >> 12) & 3;
        int K  = kc * 32 + ((e >> 3) << 4) + ((ln >> 4) << 3) + (e & 7);  // B operand: table order K/lane
        int i  = (K & 7) * 16 + (K >> 3);         // inverse of column permutation
        int n  = nt * 16 + (ln & 15);
        float v = __expf(tr[i * NL + n]);
        bf16_t h = (bf16_t)v;
        EBhi[s] = h;
        EBlo[s] = (bf16_t)(v - (float)h);
    }
    __syncthreads();

    // ---- Per-row sequence lengths (mask is a prefix: mask[b,t] = t < len) ----
    int rowB = b0 + c;
    const uint32_t* m32 = (const uint32_t*)(mask + (size_t)rowB * Tlen + halfSel * 256);
    int lsum = 0;
    #pragma unroll 8
    for (int q = 0; q < 64; ++q) {
        uint32_t v = m32[q];
        lsum += (int)((v & 0xFF) + ((v >> 8) & 0xFF) + ((v >> 16) & 0xFF) + (v >> 24));
    }
    lsum += __shfl_xor(lsum, 16, 32);   // lane holds len[row = c]

    int lenrow[8];
    int wavemax = 0;
    #pragma unroll
    for (int r = 0; r < 8; ++r) {
        lenrow[r] = __shfl(lsum, r + 8 * halfSel, 32);
        wavemax = max(wavemax, lenrow[r]);
    }
    for (int d = 16; d >= 1; d >>= 1)
        wavemax = max(wavemax, __shfl_xor(wavemax, d, 32));

    // ---- State init: Sv = exp(tr[BOS,:] + feat[:,0,:]), Mcum = 0 ----
    float Sv[8][8];       // [nt][r], WMMA C-layout
    float Mcum[8];        // per-row log scale
    #pragma unroll
    for (int r = 0; r < 8; ++r) Mcum[r] = 0.0f;
    #pragma unroll
    for (int nt = 0; nt < 8; ++nt) {
        int n = nt * 16 + c;
        float trb = tr[BOS_LAB * NL + n];
        #pragma unroll
        for (int r = 0; r < 8; ++r) {
            int b = b0 + r + 8 * halfSel;
            Sv[nt][r] = __expf(trb + feat[(size_t)b * Tlen * NL + n]);
        }
    }

    bf16_t* myPAhi = PAhi[wave];
    bf16_t* myPAlo = PAlo[wave];
    const int cstore = c * 8;

    for (int t = 1; t < wavemax; ++t) {
        // Prefetch next timestep's 16x128 feature tile (4 cachelines x 16 rows
        // covered by 2 global_prefetch instructions across the 32 lanes).
        if (t + 1 < wavemax) {
            const float* pf = feat + (size_t)(b0 + c) * Tlen * NL
                                   + (size_t)(t + 1) * NL + halfSel * 32;
            __builtin_prefetch(pf, 0, 1);
            __builtin_prefetch(pf + 64, 0, 1);
        }

        // Linear-domain row max over 128 labels (== log-domain max).
        float Mr[8], Rinv[8];
        #pragma unroll
        for (int r = 0; r < 8; ++r) {
            float mx = Sv[0][r];
            #pragma unroll
            for (int nt = 1; nt < 8; ++nt) mx = fmaxf(mx, Sv[nt][r]);
            for (int d = 1; d < 16; d <<= 1) mx = fmaxf(mx, __shfl_xor(mx, d, 32));
            Mr[r]   = mx;
            Rinv[r] = 1.0f / mx;
        }

        // P = Sv/max, split hi/lo bf16, stored with permuted cols -> one b128/row.
        #pragma unroll
        for (int r = 0; r < 8; ++r) {
            uint4 hp, lp;
            unsigned hw[4], lw[4];
            #pragma unroll
            for (int p = 0; p < 4; ++p) {
                float e0 = Sv[2 * p][r]     * Rinv[r];
                float e1 = Sv[2 * p + 1][r] * Rinv[r];
                bf16_t h0 = (bf16_t)e0, h1 = (bf16_t)e1;
                hw[p] = pack2bf(h0, h1);
                lw[p] = pack2bf((bf16_t)(e0 - (float)h0), (bf16_t)(e1 - (float)h1));
            }
            hp.x = hw[0]; hp.y = hw[1]; hp.z = hw[2]; hp.w = hw[3];
            lp.x = lw[0]; lp.y = lw[1]; lp.z = lw[2]; lp.w = lw[3];
            int m = r + 8 * halfSel;
            *(uint4*)&myPAhi[m * NL + cstore] = hp;
            *(uint4*)&myPAlo[m * NL + cstore] = lp;
        }
        asm volatile("" ::: "memory");  // keep LDS stores before A-operand loads

        // A operands: row m = c; elems 0..7 -> K = kc*32 + half*8 + e,
        //             elems 8..15 -> K = kc*32 + 16 + half*8 + e.
        v16bf Ahi[4], Alo[4];
        #pragma unroll
        for (int kc = 0; kc < 4; ++kc) {
            V16U ah, al;
            const bf16_t* bh = &myPAhi[c * NL + kc * 32 + halfSel * 8];
            const bf16_t* bl = &myPAlo[c * NL + kc * 32 + halfSel * 8];
            ah.q[0] = *(const uint4*)(bh);
            ah.q[1] = *(const uint4*)(bh + 16);
            al.q[0] = *(const uint4*)(bl);
            al.q[1] = *(const uint4*)(bl + 16);
            Ahi[kc] = ah.v;
            Alo[kc] = al.v;
        }

        // S = Phi*Ehi + Phi*Elo + Plo*Ehi, then Sv' = S * exp(feat) (masked).
        #pragma unroll
        for (int nt = 0; nt < 8; ++nt) {
            int n = nt * 16 + c;
            const float* fcol = feat + (size_t)b0 * Tlen * NL + (size_t)t * NL + n;
            // Issue feature loads unconditionally BEFORE the WMMA chain so the
            // matmul hides their latency (select applied afterwards).
            float fval[8];
            #pragma unroll
            for (int r = 0; r < 8; ++r)
                fval[r] = fcol[(size_t)(r + 8 * halfSel) * Tlen * NL];

            v8f acc = {};
            #pragma unroll
            for (int kc = 0; kc < 4; ++kc) {
                const v16bf Bh = *(const v16bf*)&EBhi[((kc * 8 + nt) * 32 + lane) * 16];
                const v16bf Bl = *(const v16bf*)&EBlo[((kc * 8 + nt) * 32 + lane) * 16];
                acc = __builtin_amdgcn_wmma_f32_16x16x32_bf16(
                          false, Alo[kc], false, Bh, (short)0, acc, false, false);
                acc = __builtin_amdgcn_wmma_f32_16x16x32_bf16(
                          false, Ahi[kc], false, Bl, (short)0, acc, false, false);
                acc = __builtin_amdgcn_wmma_f32_16x16x32_bf16(
                          false, Ahi[kc], false, Bh, (short)0, acc, false, false);
            }
            #pragma unroll
            for (int r = 0; r < 8; ++r) {
                float nsv = acc[r] * __expf(fval[r]);
                Sv[nt][r] = (t < lenrow[r]) ? nsv : Sv[nt][r];
            }
        }
        // Running log scale (one log per row instead of 64 per-element logs).
        #pragma unroll
        for (int r = 0; r < 8; ++r) {
            float nm = Mcum[r] + __logf(Mr[r]);
            Mcum[r] = (t < lenrow[r]) ? nm : Mcum[r];
        }
    }

    // ---- logZ[b] = Mcum + log( sum_j Sv_j * exp(tr[j,EOS]) ) ----
    float Ee[8];
    #pragma unroll
    for (int nt = 0; nt < 8; ++nt)
        Ee[nt] = __expf(tr[(nt * 16 + c) * NL + EOS_LAB]);

    #pragma unroll
    for (int r = 0; r < 8; ++r) {
        float s = 0.0f;
        #pragma unroll
        for (int nt = 0; nt < 8; ++nt) s += Sv[nt][r] * Ee[nt];
        for (int d = 1; d < 16; d <<= 1) s += __shfl_xor(s, d, 32);
        if (c == 0)
            logZ[b0 + r + 8 * halfSel] = Mcum[r] + __logf(s);
    }
}

// ---------------------------------------------------------------------------
// Kernel 3: out = mean(logZ - gold) = -mean(gold - logZ)
// ---------------------------------------------------------------------------
__global__ __launch_bounds__(256) void crf_reduce(
    const float* __restrict__ gold, const float* __restrict__ logZ,
    float* __restrict__ out)
{
    __shared__ float sm[256];
    int tid = threadIdx.x;
    float s = (logZ[tid] - gold[tid]) + (logZ[tid + 256] - gold[tid + 256]);
    sm[tid] = s;
    __syncthreads();
    for (int d = 128; d >= 1; d >>= 1) {
        if (tid < d) sm[tid] += sm[tid + d];
        __syncthreads();
    }
    if (tid == 0) out[0] = sm[0] / (float)Bsz;
}

extern "C" void kernel_launch(void* const* d_in, const int* in_sizes, int n_in,
                              void* d_out, int out_size, void* d_ws, size_t ws_size,
                              hipStream_t stream) {
    (void)in_sizes; (void)n_in; (void)out_size; (void)ws_size;
    const float*         feat = (const float*)d_in[0];
    const int*           tags = (const int*)d_in[1];
    const unsigned char* mask = (const unsigned char*)d_in[2];
    const float*         tr   = (const float*)d_in[3];

    float* gold = (float*)d_ws;
    float* logZ = gold + Bsz;

    crf_gold   <<<64, 256, 0, stream>>>(feat, tags, mask, tr, gold);
    crf_forward<<< 8, 128, 0, stream>>>(feat, mask, tr, logZ);
    crf_reduce <<< 1, 256, 0, stream>>>(gold, logZ, (float*)d_out);
}
